// RoboticCWRRT_28080496181591
// MI455X (gfx1250) — matrix operand order repaired and checked by reading, not compiled, not run on hardware
//
#include <hip/hip_runtime.h>
#include <math.h>

// ---------------------------------------------------------------------------
// Types for CDNA5 WMMA (wave32): v_wmma_f32_16x16x32_bf16
// ---------------------------------------------------------------------------
typedef __bf16 bf16;
typedef __attribute__((ext_vector_type(16))) __bf16 v16bf;
typedef __attribute__((ext_vector_type(8)))  __bf16 v8bf;
typedef __attribute__((ext_vector_type(8)))  float  v8f;

union V16U { v16bf v; v8bf h[2]; };

// Model constants
#define Bsz   64
#define Tlen  512
#define OBS   64
#define Edim  256
#define Hh    4
#define HD    64
#define MEM   16
#define NTHREADS 512   // 16 waves

// ---------------------------------------------------------------------------
// WMMA helpers
// ---------------------------------------------------------------------------
__device__ __forceinline__ v8f wmma_bf16(v16bf a, v16bf b, v8f c) {
  return __builtin_amdgcn_wmma_f32_16x16x32_bf16(
      /*neg_a=*/false, a, /*neg_b=*/false, b,
      /*c_mod=*/(short)0, c, /*reuse_a=*/false, /*reuse_b=*/false);
}

// Opaque zero (SGPR): added to weight pointers once per scan iteration so
// LICM cannot hoist the loop-invariant B-fragment loads (round-1 showed
// hoist-then-spill to scratch).  Unlike laundering the pointer through an
// integer, this preserves the kernarg/global provenance, so the loads stay
// global_load_b128 (LOADcnt only) instead of flat_load_b128 (LOADcnt+DScnt)
// as observed in round-2 disasm.
__device__ __forceinline__ size_t opaque_zero() {
  unsigned int x = 0;
  asm volatile("" : "+s"(x));
  return (size_t)x;
}

// A-fragment (16x32 bf16, MxK) from a row-major bf16 buffer [16][stride].
// Lane layout: lanes 0-15 M=0..15, elem i -> K = half*16 + i (contiguous,
// matching the packed B fragments).  Both halves are 8-contiguous bf16.
__device__ __forceinline__ v16bf load_a_frag(const bf16* base, int stride,
                                             int k0, int lane) {
  const int m = lane & 15;
  const int half = lane >> 4;
  const bf16* p = base + (size_t)m * stride + k0 + half * 16;
  V16U u;
  u.h[0] = *(const v8bf*)(p);
  u.h[1] = *(const v8bf*)(p + 8);
  return u.v;
}

// B-fragment (32x16 bf16, KxN) from pre-packed weights: contiguous 16 bf16
// per lane per (ntile,ktile).  Packing done by pack_kernel below.
__device__ __forceinline__ v16bf load_b_frag(const bf16* Wp, int KT,
                                             int nt, int kt, int lane) {
  return *(const v16bf*)(Wp + (((size_t)nt * KT + kt) * 32 + lane) * 16);
}

// ---------------------------------------------------------------------------
// Weight packing: f32 row-major W[K][N] -> bf16 fragments.
// element (lane,i) of tile (nt,kt):  k = kt*32 + (lane>>4)*16 + i
//                                    n = nt*16 + (lane&15)
// ---------------------------------------------------------------------------
__global__ void pack_kernel(const float* __restrict__ src,
                            bf16* __restrict__ dst, int K, int N) {
  int tid = blockIdx.x * blockDim.x + threadIdx.x;
  if (tid >= K * N) return;
  int i    = tid & 15;
  int lane = (tid >> 4) & 31;
  int tile = tid >> 9;
  int KT = K >> 5;
  int kt = tile % KT;
  int nt = tile / KT;
  int k = kt * 32 + (lane >> 4) * 16 + i;
  int n = nt * 16 + (lane & 15);
  dst[tid] = (bf16)src[(size_t)k * N + n];
}

// ---------------------------------------------------------------------------
// Recurrent scan kernel: one workgroup owns 16 batch rows for all T steps.
// grid = 4 blocks x 512 threads (16 waves).
// ---------------------------------------------------------------------------
__global__ __launch_bounds__(NTHREADS, 1)
void scan_kernel(const float* __restrict__ obs,
                 const unsigned char* __restrict__ freeze,
                 const float* __restrict__ bi,   const float* __restrict__ temb,
                 const float* __restrict__ alpha,const float* __restrict__ lam,
                 const float* __restrict__ ln1s, const float* __restrict__ ln1b,
                 const float* __restrict__ bq,   const float* __restrict__ bk,
                 const float* __restrict__ bv,   const float* __restrict__ bo,
                 const float* __restrict__ ln2s, const float* __restrict__ ln2b,
                 const float* __restrict__ b1,   const float* __restrict__ b2,
                 const bf16* __restrict__ WiP,  const bf16* __restrict__ WqP,
                 const bf16* __restrict__ WkP,  const bf16* __restrict__ WvP,
                 const bf16* __restrict__ WoP,  const bf16* __restrict__ W1P,
                 const bf16* __restrict__ W2P,
                 bf16* __restrict__ xoutP,
                 float* __restrict__ kring_g, float* __restrict__ vring_g) {
  __shared__ float xbuf[16][Edim];            // x / x_in / x_mid / x_out (f32)
  __shared__ float ssum[16][Edim];            // recurrent sum (f32)
  __shared__ __align__(32) bf16 abuf[16][Edim];   // GEMM A staging
  __shared__ __align__(32) bf16 hbuf[16][Edim];   // FFN chunk staging
  __shared__ __align__(32) bf16 qbuf[16][Edim];   // q (scaled)
  __shared__ float scores[16][Hh][MEM];
  __shared__ float sa[Edim], sl[Edim];        // sigmoid(alpha), sigmoid(lam)

  const int tid  = threadIdx.x;
  const int lane = tid & 31;
  const int wave = tid >> 5;
  const int bb   = blockIdx.x * 16;           // first batch row of this WG

  float* kr = kring_g + (size_t)blockIdx.x * (MEM * 16 * Edim);
  float* vr = vring_g + (size_t)blockIdx.x * (MEM * 16 * Edim);

  // ---- init: gates, ssum=0, ring caches = bias (k/v of zero mem rows) ----
  if (tid < Edim) {
    sa[tid] = 1.f / (1.f + expf(-alpha[tid]));
    sl[tid] = 1.f / (1.f + expf(-lam[tid]));
  }
  for (int idx = tid; idx < 16 * Edim; idx += NTHREADS)
    ssum[idx >> 8][idx & 255] = 0.f;
  for (int idx = tid; idx < MEM * 16 * Edim; idx += NTHREADS) {
    int col = idx & 255;
    kr[idx] = bk[col];
    vr[idx] = bv[col];
  }
  __threadfence();
  __syncthreads();

  const float qscale = 0.125f;                // 1/sqrt(HD)

  for (int t = 0; t < Tlen; ++t) {
    // Per-iteration opaque +0 keeps B-fragment loads inside the loop
    // (L2 hits each step) without losing the global address space.
    const size_t zz = opaque_zero();
    const bf16* WiQ = WiP + zz;
    const bf16* WqQ = WqP + zz;
    const bf16* WkQ = WkP + zz;
    const bf16* WvQ = WvP + zz;
    const bf16* WoQ = WoP + zz;
    const bf16* W1Q = W1P + zz;
    const bf16* W2Q = W2P + zz;

    // ---- 1. stage obs tile (16 x 64) -> abuf (bf16) -------------------
#pragma unroll
    for (int idx = tid; idx < 16 * OBS; idx += NTHREADS) {
      int r = idx >> 6, f = idx & 63;
      abuf[r][f] = (bf16)obs[((size_t)(bb + r) * Tlen + t) * OBS + f];
    }
    __syncthreads();

    // ---- 2. input projection: x = obs @ Wi  (K=64, N=256) -------------
    {
      v8f acc = {};
#pragma unroll
      for (int kt = 0; kt < 2; ++kt) {
        v16bf a = load_a_frag(&abuf[0][0], Edim, kt * 32, lane);
        v16bf b = load_b_frag(WiQ, 2, wave, kt, lane);
        acc = wmma_bf16(a, b, acc);
      }
      int n = wave * 16 + (lane & 15), half = lane >> 4;
#pragma unroll
      for (int j = 0; j < 8; ++j) xbuf[j + 8 * half][n] = acc[j];
    }
    __syncthreads();

    // ---- 3. x_in = x + bi + temb[t%10] + ssum * sigm(alpha) -----------
    const float* te = temb + (size_t)(t % 10) * Edim;
#pragma unroll
    for (int idx = tid; idx < 16 * Edim; idx += NTHREADS) {
      int r = idx >> 8, e = idx & 255;
      float x = xbuf[r][e] + bi[e] + te[e] + ssum[r][e] * sa[e];
      xbuf[r][e] = x;
      abuf[r][e] = (bf16)x;                    // A input for k/v of new entry
    }
    __syncthreads();

    // ---- 4. k/v for the NEW memory entry only (ring-buffer cache) -----
    {
      const int slot = t & (MEM - 1);
      v8f ak = {}, av = {};
#pragma unroll
      for (int kt = 0; kt < 8; ++kt) {
        v16bf a = load_a_frag(&abuf[0][0], Edim, kt * 32, lane);
        ak = wmma_bf16(a, load_b_frag(WkQ, 8, wave, kt, lane), ak);
        av = wmma_bf16(a, load_b_frag(WvQ, 8, wave, kt, lane), av);
      }
      int n = wave * 16 + (lane & 15), half = lane >> 4;
#pragma unroll
      for (int j = 0; j < 8; ++j) {
        int m = j + 8 * half;
        kr[slot * (16 * Edim) + m * Edim + n] = ak[j] + bk[n];
        vr[slot * (16 * Edim) + m * Edim + n] = av[j] + bv[n];
      }
    }
    __threadfence();       // make ring writes visible to whole WG
    __syncthreads();

    // ---- 5. LN1 (wave r handles row r, full-wave shuffle reduction) ---
    {
      int r = wave;
      float vals[8], s = 0.f;
#pragma unroll
      for (int q = 0; q < 8; ++q) { vals[q] = xbuf[r][lane * 8 + q]; s += vals[q]; }
      for (int m = 16; m; m >>= 1) s += __shfl_xor(s, m);
      float mean = s * (1.f / 256.f), vv = 0.f;
#pragma unroll
      for (int q = 0; q < 8; ++q) { float d = vals[q] - mean; vv += d * d; }
      for (int m = 16; m; m >>= 1) vv += __shfl_xor(vv, m);
      float inv = rsqrtf(vv * (1.f / 256.f) + 1e-6f);
#pragma unroll
      for (int q = 0; q < 8; ++q) {
        int c = lane * 8 + q;
        abuf[r][c] = (bf16)((vals[q] - mean) * inv * ln1s[c] + ln1b[c]);
      }
    }
    __syncthreads();

    // ---- 6. q = (y @ Wq + bq) * scale ---------------------------------
    {
      v8f acc = {};
#pragma unroll
      for (int kt = 0; kt < 8; ++kt)
        acc = wmma_bf16(load_a_frag(&abuf[0][0], Edim, kt * 32, lane),
                        load_b_frag(WqQ, 8, wave, kt, lane), acc);
      int n = wave * 16 + (lane & 15), half = lane >> 4;
#pragma unroll
      for (int j = 0; j < 8; ++j)
        qbuf[j + 8 * half][n] = (bf16)((acc[j] + bq[n]) * qscale);
    }
    __syncthreads();

    // ---- 7. attention scores q . k (per row, head, slot) --------------
#pragma unroll
    for (int task = tid; task < 16 * Hh * MEM; task += NTHREADS) {
      int r = task >> 6, h = (task >> 4) & 3, m = task & 15;
      const float* kp = kr + m * (16 * Edim) + r * Edim + h * HD;
      float s = 0.f;
#pragma unroll
      for (int d = 0; d < HD; ++d) s += (float)qbuf[r][h * HD + d] * kp[d];
      scores[r][h][m] = s;
    }
    __syncthreads();

    // ---- 8. softmax over MEM ------------------------------------------
    if (tid < 16 * Hh) {
      int r = tid >> 2, h = tid & 3;
      float mx = -1e30f;
#pragma unroll
      for (int m = 0; m < MEM; ++m) mx = fmaxf(mx, scores[r][h][m]);
      float sum = 0.f;
#pragma unroll
      for (int m = 0; m < MEM; ++m) {
        float e = expf(scores[r][h][m] - mx);
        scores[r][h][m] = e; sum += e;
      }
      float inv = 1.f / sum;
#pragma unroll
      for (int m = 0; m < MEM; ++m) scores[r][h][m] *= inv;
    }
    __syncthreads();

    // ---- 9. ctx = attn @ v  -> abuf (bf16) ----------------------------
#pragma unroll
    for (int idx = tid; idx < 16 * Edim; idx += NTHREADS) {
      int r = idx >> 8, c = idx & 255, h = c >> 6;
      float s = 0.f;
#pragma unroll
      for (int m = 0; m < MEM; ++m)
        s += scores[r][h][m] * vr[m * (16 * Edim) + r * Edim + c];
      abuf[r][c] = (bf16)s;
    }
    __syncthreads();

    // ---- 10. out proj: x_mid = x_in + ctx @ Wo + bo -------------------
    {
      v8f acc = {};
#pragma unroll
      for (int kt = 0; kt < 8; ++kt)
        acc = wmma_bf16(load_a_frag(&abuf[0][0], Edim, kt * 32, lane),
                        load_b_frag(WoQ, 8, wave, kt, lane), acc);
      int n = wave * 16 + (lane & 15), half = lane >> 4;
#pragma unroll
      for (int j = 0; j < 8; ++j) xbuf[j + 8 * half][n] += acc[j] + bo[n];
    }
    __syncthreads();

    // ---- 11. LN2 -> abuf ----------------------------------------------
    {
      int r = wave;
      float vals[8], s = 0.f;
#pragma unroll
      for (int q = 0; q < 8; ++q) { vals[q] = xbuf[r][lane * 8 + q]; s += vals[q]; }
      for (int m = 16; m; m >>= 1) s += __shfl_xor(s, m);
      float mean = s * (1.f / 256.f), vv = 0.f;
#pragma unroll
      for (int q = 0; q < 8; ++q) { float d = vals[q] - mean; vv += d * d; }
      for (int m = 16; m; m >>= 1) vv += __shfl_xor(vv, m);
      float inv = rsqrtf(vv * (1.f / 256.f) + 1e-6f);
#pragma unroll
      for (int q = 0; q < 8; ++q) {
        int c = lane * 8 + q;
        abuf[r][c] = (bf16)((vals[q] - mean) * inv * ln2s[c] + ln2b[c]);
      }
    }
    __syncthreads();

    // ---- 12. FFN, K-chunked so 16x1024 intermediate stays in regs/LDS -
    {
      v8f accF = {};                            // FFN2 output tile (this wave)
      const int nl = lane & 15, half = lane >> 4;
#pragma unroll
      for (int c = 0; c < 4; ++c) {
        v8f acc1 = {};
#pragma unroll
        for (int kt = 0; kt < 8; ++kt)          // FFN1: y @ W1[:, c*256+...]
          acc1 = wmma_bf16(load_a_frag(&abuf[0][0], Edim, kt * 32, lane),
                           load_b_frag(W1Q, 8, c * 16 + wave, kt, lane), acc1);
        int gcol = (c * 16 + wave) * 16 + nl;   // global W1 output column
#pragma unroll
        for (int j = 0; j < 8; ++j) {           // gelu (tanh approx)
          float x = acc1[j] + b1[gcol];
          float g = 0.5f * x *
                    (1.f + tanhf(0.7978845608f * (x + 0.044715f * x * x * x)));
          hbuf[j + 8 * half][wave * 16 + nl] = (bf16)g;
        }
        __syncthreads();                        // hbuf chunk ready
#pragma unroll
        for (int kt = 0; kt < 8; ++kt)          // FFN2 partial over this chunk
          accF = wmma_bf16(load_a_frag(&hbuf[0][0], Edim, kt * 32, lane),
                           load_b_frag(W2Q, 32, wave, c * 8 + kt, lane), accF);
        __syncthreads();                        // before hbuf is overwritten
      }
      int n = wave * 16 + nl;
#pragma unroll
      for (int j = 0; j < 8; ++j) xbuf[j + 8 * half][n] += accF[j] + b2[n];
    }
    __syncthreads();

    // ---- 13. store x_out (bf16) + ssum update with freeze gate --------
#pragma unroll
    for (int idx = tid; idx < 16 * Edim; idx += NTHREADS) {
      int r = idx >> 8, e = idx & 255;
      float xo = xbuf[r][e];
      xoutP[((size_t)(bb + r) * Tlen + t) * Edim + e] = (bf16)xo;
      float l  = sl[e];
      float ns = ssum[r][e] * l + xo * (1.f - l);
      if (!freeze[(size_t)(bb + r) * Tlen + t]) ssum[r][e] = ns;
    }
    __syncthreads();
  }
}

// ---------------------------------------------------------------------------
// Output head: logits[row][n] = xout[row][:] @ Wa + ba,  row = b*T + t.
// grid = 2048 blocks (16-row M tiles), 512 threads; wave owns 4 N-tiles.
// ---------------------------------------------------------------------------
__global__ __launch_bounds__(NTHREADS)
void head_kernel(const bf16* __restrict__ xoutP, const bf16* __restrict__ WaP,
                 const float* __restrict__ ba, float* __restrict__ out) {
  const int lane = threadIdx.x & 31;
  const int wave = threadIdx.x >> 5;
  const size_t mt = blockIdx.x;
  const bf16* arow = xoutP + mt * 16 * Edim;

  v8f acc[4] = {{}, {}, {}, {}};
#pragma unroll
  for (int kt = 0; kt < 8; ++kt) {
    v16bf a = load_a_frag(arow, Edim, kt * 32, lane);
#pragma unroll
    for (int q = 0; q < 4; ++q)
      acc[q] = wmma_bf16(a, load_b_frag(WaP, 8, wave * 4 + q, kt, lane), acc[q]);
  }
  const int nl = lane & 15, half = lane >> 4;
#pragma unroll
  for (int q = 0; q < 4; ++q) {
    int n = (wave * 4 + q) * 16 + nl;
    float bias = ba[n];
#pragma unroll
    for (int j = 0; j < 8; ++j) {
      size_t row = mt * 16 + j + 8 * half;
      out[row * 1024 + n] = acc[q][j] + bias;
    }
  }
}

// ---------------------------------------------------------------------------
// Host launcher
// ---------------------------------------------------------------------------
extern "C" void kernel_launch(void* const* d_in, const int* in_sizes, int n_in,
                              void* d_out, int out_size, void* d_ws,
                              size_t ws_size, hipStream_t stream) {
  (void)in_sizes; (void)n_in; (void)out_size; (void)ws_size;

  const float* obs    = (const float*)d_in[0];
  const unsigned char* frz = (const unsigned char*)d_in[1];
  const float* Wi   = (const float*)d_in[2];
  const float* bi   = (const float*)d_in[3];
  const float* temb = (const float*)d_in[4];
  const float* alpha= (const float*)d_in[5];
  const float* lam  = (const float*)d_in[6];
  const float* ln1s = (const float*)d_in[7];
  const float* ln1b = (const float*)d_in[8];
  const float* Wq   = (const float*)d_in[9];
  const float* bq   = (const float*)d_in[10];
  const float* Wk   = (const float*)d_in[11];
  const float* bk   = (const float*)d_in[12];
  const float* Wv   = (const float*)d_in[13];
  const float* bv   = (const float*)d_in[14];
  const float* Wo   = (const float*)d_in[15];
  const float* bo   = (const float*)d_in[16];
  const float* ln2s = (const float*)d_in[17];
  const float* ln2b = (const float*)d_in[18];
  const float* W1   = (const float*)d_in[19];
  const float* b1   = (const float*)d_in[20];
  const float* W2   = (const float*)d_in[21];
  const float* b2   = (const float*)d_in[22];
  const float* Wa   = (const float*)d_in[23];
  const float* ba   = (const float*)d_in[24];

  char* ws = (char*)d_ws;
  size_t off = 0;
  auto carve = [&](size_t bytes) -> void* {
    void* p = ws + off;
    off += (bytes + 255) & ~(size_t)255;
    return p;
  };
  bf16* WiP = (bf16*)carve((size_t)OBS * Edim * 2);
  bf16* WqP = (bf16*)carve((size_t)Edim * Edim * 2);
  bf16* WkP = (bf16*)carve((size_t)Edim * Edim * 2);
  bf16* WvP = (bf16*)carve((size_t)Edim * Edim * 2);
  bf16* WoP = (bf16*)carve((size_t)Edim * Edim * 2);
  bf16* W1P = (bf16*)carve((size_t)Edim * 4 * Edim * 2);
  bf16* W2P = (bf16*)carve((size_t)4 * Edim * Edim * 2);
  bf16* WaP = (bf16*)carve((size_t)Edim * 1024 * 2);
  bf16* xoutP = (bf16*)carve((size_t)Bsz * Tlen * Edim * 2);
  float* kring = (float*)carve((size_t)4 * MEM * 16 * Edim * 4);
  float* vring = (float*)carve((size_t)4 * MEM * 16 * Edim * 4);

  // Pack all weights into WMMA B-fragment layout (bf16).
  auto pack = [&](const float* src, bf16* dst, int K, int N) {
    int total = K * N;
    pack_kernel<<<(total + 255) / 256, 256, 0, stream>>>(src, dst, K, N);
  };
  pack(Wi, WiP, OBS, Edim);
  pack(Wq, WqP, Edim, Edim);     // (E, H*HD) row-major
  pack(Wk, WkP, Edim, Edim);
  pack(Wv, WvP, Edim, Edim);
  pack(Wo, WoP, Edim, Edim);     // (H*HD, E) row-major
  pack(W1, W1P, Edim, 4 * Edim);
  pack(W2, W2P, 4 * Edim, Edim);
  pack(Wa, WaP, Edim, 1024);

  // Recurrent scan: 4 workgroups x 16 batch rows, 512 sequential steps.
  scan_kernel<<<Bsz / 16, NTHREADS, 0, stream>>>(
      obs, frz, bi, temb, alpha, lam, ln1s, ln1b, bq, bk, bv, bo, ln2s, ln2b,
      b1, b2, WiP, WqP, WkP, WvP, WoP, W1P, W2P, xoutP, kring, vring);

  // Output head: (B*T, 256) @ (256, 1024).
  head_kernel<<<(Bsz * Tlen) / 16, NTHREADS, 0, stream>>>(
      xoutP, WaP, ba, (float*)d_out);
}
